// TransformerBlock_46256797778412
// MI455X (gfx1250) — matrix operand order repaired and checked by reading, not compiled, not run on hardware
//
#include <hip/hip_runtime.h>
#include <hip/hip_bf16.h>
#include <math.h>

// ---------------------------------------------------------------------------
// Transformer block (B=8, S=1024, D=768, FF=3072), fp32 I/O.
// All GEMMs: bf16 inputs (converted once), f32 WMMA accumulation via
// v_wmma_f32_16x16x32_bf16 (CDNA5 / gfx1250, wave32).
// Staging via GLOBAL_LOAD_ASYNC_TO_LDS (ASYNCcnt) when available.
// ---------------------------------------------------------------------------

#define TOKENS   8192      // 8 * 1024
#define SEQ      1024
#define BATCH    8
#define EMBED    768
#define FFDIM    3072
#define LN_EPS   1e-5f

// LDS tile geometry: BM=BN=128, BK=64, row stride 72 bf16 (144 B, 16B-aligned,
// conflict-free row rotation across 64 banks).
#define LDS_STRIDE 72

#if defined(__has_builtin)
#  if __has_builtin(__builtin_amdgcn_global_load_async_to_lds_b128)
#    define USE_ASYNC_LDS 1
#  endif
#  if __has_builtin(__builtin_amdgcn_s_wait_asynccnt)
#    define HAVE_WAIT_ASYNC_BUILTIN 1
#  endif
#endif
#ifndef USE_ASYNC_LDS
#  define USE_ASYNC_LDS 0
#endif
#ifndef HAVE_WAIT_ASYNC_BUILTIN
#  define HAVE_WAIT_ASYNC_BUILTIN 0
#endif

typedef __attribute__((ext_vector_type(16))) __bf16 v16bf;
typedef __attribute__((ext_vector_type(8)))  float  v8f;
typedef int v4i __attribute__((vector_size(4 * sizeof(int))));

union FragBF { v16bf v; unsigned int u[8]; };

#if USE_ASYNC_LDS
typedef __attribute__((address_space(1))) v4i* gv4i_p;   // global int4*
typedef __attribute__((address_space(3))) v4i* lv4i_p;   // LDS int4*
#endif

__device__ __forceinline__ unsigned short f32_to_bf16(float f) {
    unsigned int u = __float_as_uint(f);
    u += 0x7FFFu + ((u >> 16) & 1u);   // round-to-nearest-even
    return (unsigned short)(u >> 16);
}

__device__ __forceinline__ void wait_async_zero() {
#if USE_ASYNC_LDS
#  if HAVE_WAIT_ASYNC_BUILTIN
    __builtin_amdgcn_s_wait_asynccnt(0);
#  else
    asm volatile("s_wait_asynccnt 0x0" ::: "memory");
#  endif
#endif
}

// ---------------------------------------------------------------------------
// Elementwise f32 -> bf16 convert
// ---------------------------------------------------------------------------
__global__ void convert_bf16_kernel(const float* __restrict__ in,
                                    unsigned short* __restrict__ out, int n) {
    int i = blockIdx.x * blockDim.x + threadIdx.x;
    if (i < n) out[i] = f32_to_bf16(in[i]);
}

// ---------------------------------------------------------------------------
// Tiled transpose + convert: in f32 [R][C] -> out bf16 [C][R], batched via z.
// ---------------------------------------------------------------------------
__global__ __launch_bounds__(256)
void transpose_conv_kernel(const float* __restrict__ in,
                           unsigned short* __restrict__ out,
                           int R, int C, long inBatchStride, long outBatchStride) {
    __shared__ float tile[32][33];
    const float* src = in + (size_t)blockIdx.z * inBatchStride;
    unsigned short* dst = out + (size_t)blockIdx.z * outBatchStride;
    const int c0 = blockIdx.x * 32, r0 = blockIdx.y * 32;
    for (int i = threadIdx.y; i < 32; i += 8)
        tile[i][threadIdx.x] = src[(size_t)(r0 + i) * C + c0 + threadIdx.x];
    __syncthreads();
    for (int i = threadIdx.y; i < 32; i += 8)
        dst[(size_t)(c0 + i) * R + r0 + threadIdx.x] = f32_to_bf16(tile[threadIdx.x][i]);
}

// ---------------------------------------------------------------------------
// GEMM: C[M,N] = scale * (A[M,K] @ B[K,N]) + bias, B given transposed: Bt[N][K].
// A, Bt bf16 row-major (k contiguous). Outputs: optional f32 and/or bf16.
// BM=BN=128, BK=64, 256 threads (8 waves); wave -> 64x32 patch (4x2 WMMA
// tiles), 16 WMMAs per staging round.
// ---------------------------------------------------------------------------
__global__ __launch_bounds__(256)
void gemm_bf16_wmma(const unsigned short* __restrict__ A, long strideA, int lda,
                    const unsigned short* __restrict__ Bt, long strideB, int ldb,
                    float* Cf, unsigned short* Cb, long strideC, int ldc,
                    const float* __restrict__ bias, float scale, int K, int gelu) {
    __shared__ unsigned short As[128 * LDS_STRIDE];
    __shared__ unsigned short Bs[128 * LDS_STRIDE];

    const int tid  = threadIdx.x;
    const int lane = tid & 31;
    const int wave = tid >> 5;
    const int lhalf = lane >> 4;   // 0: lanes 0-15, 1: lanes 16-31
    const int lm    = lane & 15;

    const int m0 = blockIdx.y * 128;
    const int n0 = blockIdx.x * 128;
    const int wmOff = (wave >> 2) * 64;   // 2 waves along M
    const int wnOff = (wave & 3) * 32;    // 4 waves along N

    A  += (size_t)blockIdx.z * strideA;
    Bt += (size_t)blockIdx.z * strideB;

    const int srow = tid >> 1;           // 0..127 staging row
    const int skq  = (tid & 1) * 32;     // 0 or 32 (bf16 elems)

    v8f acc[4][2];
    const v8f vzero = {0.f, 0.f, 0.f, 0.f, 0.f, 0.f, 0.f, 0.f};
#pragma unroll
    for (int i = 0; i < 4; ++i)
#pragma unroll
        for (int j = 0; j < 2; ++j) acc[i][j] = vzero;

    for (int k0 = 0; k0 < K; k0 += 64) {
        __syncthreads();
        {
            const unsigned short* gA = A  + (size_t)(m0 + srow) * lda + k0 + skq;
            const unsigned short* gB = Bt + (size_t)(n0 + srow) * ldb + k0 + skq;
#if USE_ASYNC_LDS
            // gfx1250 async copy: global -> LDS, no VGPR round trip, ASYNCcnt.
            gv4i_p ga = (gv4i_p)gA;
            gv4i_p gb = (gv4i_p)gB;
            lv4i_p la = (lv4i_p)&As[srow * LDS_STRIDE + skq];
            lv4i_p lb = (lv4i_p)&Bs[srow * LDS_STRIDE + skq];
#pragma unroll
            for (int q = 0; q < 4; ++q) {
                __builtin_amdgcn_global_load_async_to_lds_b128(ga + q, la + q, 0, 0);
                __builtin_amdgcn_global_load_async_to_lds_b128(gb + q, lb + q, 0, 0);
            }
#else
            const uint4* ga = reinterpret_cast<const uint4*>(gA);
            const uint4* gb = reinterpret_cast<const uint4*>(gB);
            uint4 a0 = ga[0], a1 = ga[1], a2 = ga[2], a3 = ga[3];
            uint4 b0 = gb[0], b1 = gb[1], b2 = gb[2], b3 = gb[3];
            uint4* sa = reinterpret_cast<uint4*>(&As[srow * LDS_STRIDE + skq]);
            uint4* sb = reinterpret_cast<uint4*>(&Bs[srow * LDS_STRIDE + skq]);
            sa[0] = a0; sa[1] = a1; sa[2] = a2; sa[3] = a3;
            sb[0] = b0; sb[1] = b1; sb[2] = b2; sb[3] = b3;
#endif
            if (k0 + 64 < K) {  // global_prefetch_b8 of next K tile
                __builtin_prefetch(gA + 64, 0, 1);
                __builtin_prefetch(gB + 64, 0, 1);
            }
        }
        wait_async_zero();
        __syncthreads();

        // Two K=32 WMMA steps out of the staged K=64 tile.
#pragma unroll
        for (int ks = 0; ks < 64; ks += 32) {
            // Fragment loads matching CDNA5 16-bit WMMA layouts:
            //  A 16x32: lane m = lane%16; VGPR v: K = (v&3)*2 + (v>>2)*16 + half*8
            //  B 32x16: lane n = lane%16; VGPR v: K = (v&3)*2 + (v>>2)*16 + half*8
            FragBF afr[4], bfr[2];
#pragma unroll
            for (int mt = 0; mt < 4; ++mt) {
                const unsigned short* ar =
                    &As[(wmOff + mt * 16 + lm) * LDS_STRIDE + ks];
#pragma unroll
                for (int v = 0; v < 8; ++v) {
                    const int kk = ((v & 3) * 2) + ((v >> 2) * 16) + lhalf * 8;
                    afr[mt].u[v] = *reinterpret_cast<const unsigned int*>(ar + kk);
                }
            }
#pragma unroll
            for (int nt = 0; nt < 2; ++nt) {
                const unsigned short* br =
                    &Bs[(wnOff + nt * 16 + lm) * LDS_STRIDE + ks];
#pragma unroll
                for (int v = 0; v < 8; ++v) {
                    const int kk = ((v & 3) * 2) + ((v >> 2) * 16) + lhalf * 8;
                    bfr[nt].u[v] = *reinterpret_cast<const unsigned int*>(br + kk);
                }
            }
#pragma unroll
            for (int mt = 0; mt < 4; ++mt)
#pragma unroll
                for (int nt = 0; nt < 2; ++nt)
                    acc[mt][nt] = __builtin_amdgcn_wmma_f32_16x16x32_bf16(
                        false, afr[mt].v, false, bfr[nt].v,
                        (short)0, acc[mt][nt], false, false);
        }
    }

    float* cf = Cf ? Cf + (size_t)blockIdx.z * strideC : (float*)0;
    unsigned short* cb = Cb ? Cb + (size_t)blockIdx.z * strideC : (unsigned short*)0;

    // C/D layout: VGPR r -> M = r + 8*half (lane half), N = lane%16
#pragma unroll
    for (int mt = 0; mt < 4; ++mt) {
#pragma unroll
        for (int nt = 0; nt < 2; ++nt) {
            const int n = n0 + wnOff + nt * 16 + lm;
            const float bv = bias ? bias[n] : 0.0f;
#pragma unroll
            for (int r = 0; r < 8; ++r) {
                const int m = m0 + wmOff + mt * 16 + lhalf * 8 + r;
                float v = acc[mt][nt][r] * scale + bv;
                if (gelu) v = 0.5f * v * (1.0f + erff(v * 0.70710678118654752f));
                const size_t off = (size_t)m * ldc + n;
                if (cf) cf[off] = v;
                if (cb) cb[off] = f32_to_bf16(v);
            }
        }
    }
}

// ---------------------------------------------------------------------------
// Row softmax: f32 scores [rows][1024] -> bf16 probabilities
// ---------------------------------------------------------------------------
__global__ __launch_bounds__(256)
void softmax_kernel(const float* __restrict__ S, unsigned short* __restrict__ P,
                    int cols) {
    __shared__ float red[256];
    const int row = blockIdx.x;
    const int tid = threadIdx.x;
    const float* src = S + (size_t)row * cols;
    unsigned short* dst = P + (size_t)row * cols;

    float lmax = -3.402823466e38f;
    for (int c = tid; c < cols; c += 256) lmax = fmaxf(lmax, src[c]);
    red[tid] = lmax; __syncthreads();
    for (int s = 128; s > 0; s >>= 1) {
        if (tid < s) red[tid] = fmaxf(red[tid], red[tid + s]);
        __syncthreads();
    }
    const float mx = red[0]; __syncthreads();

    float ev[4];
    float lsum = 0.f;
    int idx = 0;
    for (int c = tid; c < cols; c += 256) {
        float e = expf(src[c] - mx);
        ev[idx++] = e; lsum += e;
    }
    red[tid] = lsum; __syncthreads();
    for (int s = 128; s > 0; s >>= 1) {
        if (tid < s) red[tid] += red[tid + s];
        __syncthreads();
    }
    const float inv = 1.0f / red[0];
    idx = 0;
    for (int c = tid; c < cols; c += 256) dst[c] = f32_to_bf16(ev[idx++] * inv);
}

// ---------------------------------------------------------------------------
// out = xres + LayerNorm(y) * g + b   (row length = cols; ddof=0 variance)
// Optional f32 and bf16 outputs.
// ---------------------------------------------------------------------------
__global__ __launch_bounds__(256)
void ln_residual_kernel(const float* __restrict__ xres, const float* __restrict__ y,
                        const float* __restrict__ g, const float* __restrict__ b,
                        float* outf, unsigned short* outb, int cols) {
    __shared__ float red[256];
    __shared__ float red2[256];
    const int row = blockIdx.x;
    const int tid = threadIdx.x;
    const float* yr = y + (size_t)row * cols;
    const float* xr = xres + (size_t)row * cols;

    float s = 0.f, s2 = 0.f;
    for (int c = tid; c < cols; c += 256) {
        float v = yr[c];
        s += v; s2 += v * v;
    }
    red[tid] = s; red2[tid] = s2; __syncthreads();
    for (int st = 128; st > 0; st >>= 1) {
        if (tid < st) { red[tid] += red[tid + st]; red2[tid] += red2[tid + st]; }
        __syncthreads();
    }
    const float mu  = red[0] / (float)cols;
    const float var = red2[0] / (float)cols - mu * mu;
    const float rstd = rsqrtf(var + LN_EPS);

    for (int c = tid; c < cols; c += 256) {
        float v = xr[c] + (yr[c] - mu) * rstd * g[c] + b[c];
        if (outf) outf[(size_t)row * cols + c] = v;
        if (outb) outb[(size_t)row * cols + c] = f32_to_bf16(v);
    }
}

// ---------------------------------------------------------------------------
// Launch
// ---------------------------------------------------------------------------
extern "C" void kernel_launch(void* const* d_in, const int* in_sizes, int n_in,
                              void* d_out, int out_size, void* d_ws, size_t ws_size,
                              hipStream_t stream) {
    const float* x    = (const float*)d_in[0];
    const float* Wq   = (const float*)d_in[1];
    const float* bq   = (const float*)d_in[2];
    const float* Wk   = (const float*)d_in[3];
    const float* bk   = (const float*)d_in[4];
    const float* Wv   = (const float*)d_in[5];
    const float* bv   = (const float*)d_in[6];
    const float* W1   = (const float*)d_in[7];
    const float* b1   = (const float*)d_in[8];
    const float* W2   = (const float*)d_in[9];
    const float* b2   = (const float*)d_in[10];
    const float* g1   = (const float*)d_in[11];
    const float* bt1  = (const float*)d_in[12];
    const float* g2   = (const float*)d_in[13];
    const float* bt2  = (const float*)d_in[14];
    float* out = (float*)d_out;

    // --- workspace carve (256B aligned slices) ---
    char* p = (char*)d_ws;
    auto carve = [&](size_t bytes) -> void* {
        void* r = (void*)p;
        p += (bytes + 255) & ~(size_t)255;
        return r;
    };
    unsigned short* xb   = (unsigned short*)carve((size_t)TOKENS * EMBED * 2);
    unsigned short* Wqt  = (unsigned short*)carve((size_t)EMBED * EMBED * 2);
    unsigned short* Wkt  = (unsigned short*)carve((size_t)EMBED * EMBED * 2);
    unsigned short* Wvt  = (unsigned short*)carve((size_t)EMBED * EMBED * 2);
    unsigned short* W1t  = (unsigned short*)carve((size_t)FFDIM * EMBED * 2);
    unsigned short* W2t  = (unsigned short*)carve((size_t)EMBED * FFDIM * 2);
    unsigned short* Qb   = (unsigned short*)carve((size_t)TOKENS * EMBED * 2);
    unsigned short* Kb   = (unsigned short*)carve((size_t)TOKENS * EMBED * 2);
    float*          Vf   = (float*)carve((size_t)TOKENS * EMBED * 4);
    unsigned short* Vt   = (unsigned short*)carve((size_t)BATCH * EMBED * SEQ * 2);
    float*          scores = (float*)carve((size_t)BATCH * SEQ * SEQ * 4);
    unsigned short* attnb  = (unsigned short*)carve((size_t)BATCH * SEQ * SEQ * 2);
    float*          attno  = (float*)carve((size_t)TOKENS * EMBED * 4);
    float*          x1     = (float*)carve((size_t)TOKENS * EMBED * 4);
    unsigned short* x1b    = (unsigned short*)carve((size_t)TOKENS * EMBED * 2);
    unsigned short* hb     = (unsigned short*)carve((size_t)TOKENS * FFDIM * 2);
    float*          ffn    = (float*)carve((size_t)TOKENS * EMBED * 4);

    const long SD = (long)SEQ * EMBED;     // per-batch stride of [S,D] tensors
    const long SS = (long)SEQ * SEQ;       // per-batch stride of [S,S] tensors
    const float inv_sqrt_d = 0.03608439182435161f;   // 768^-0.5

    // 1) x -> bf16
    {
        int n = TOKENS * EMBED;
        convert_bf16_kernel<<<dim3((n + 255) / 256), dim3(256), 0, stream>>>(x, xb, n);
    }
    // 2) weights -> bf16, transposed to [N][K]
    transpose_conv_kernel<<<dim3(EMBED/32, EMBED/32, 1), dim3(32, 8), 0, stream>>>(
        Wq, Wqt, EMBED, EMBED, 0, 0);
    transpose_conv_kernel<<<dim3(EMBED/32, EMBED/32, 1), dim3(32, 8), 0, stream>>>(
        Wk, Wkt, EMBED, EMBED, 0, 0);
    transpose_conv_kernel<<<dim3(EMBED/32, EMBED/32, 1), dim3(32, 8), 0, stream>>>(
        Wv, Wvt, EMBED, EMBED, 0, 0);
    transpose_conv_kernel<<<dim3(FFDIM/32, EMBED/32, 1), dim3(32, 8), 0, stream>>>(
        W1, W1t, EMBED, FFDIM, 0, 0);
    transpose_conv_kernel<<<dim3(EMBED/32, FFDIM/32, 1), dim3(32, 8), 0, stream>>>(
        W2, W2t, FFDIM, EMBED, 0, 0);

    // 3) Q = x@Wq + bq (bf16), K = x@Wk + bk (bf16), V = x@Wv + bv (f32)
    dim3 gQKV(EMBED/128, TOKENS/128, 1);
    gemm_bf16_wmma<<<gQKV, 256, 0, stream>>>(xb, 0, EMBED, Wqt, 0, EMBED,
        (float*)0, Qb, 0, EMBED, bq, 1.0f, EMBED, 0);
    gemm_bf16_wmma<<<gQKV, 256, 0, stream>>>(xb, 0, EMBED, Wkt, 0, EMBED,
        (float*)0, Kb, 0, EMBED, bk, 1.0f, EMBED, 0);
    gemm_bf16_wmma<<<gQKV, 256, 0, stream>>>(xb, 0, EMBED, Wvt, 0, EMBED,
        Vf, (unsigned short*)0, 0, EMBED, bv, 1.0f, EMBED, 0);

    // 4) V -> Vt bf16 [batch][D][S]
    transpose_conv_kernel<<<dim3(EMBED/32, SEQ/32, BATCH), dim3(32, 8), 0, stream>>>(
        Vf, Vt, SEQ, EMBED, SD, SD);

    // 5) scores = (Q @ K^T) * d^-0.5, batched; Bt = K row-major = K^T pre-transposed
    gemm_bf16_wmma<<<dim3(SEQ/128, SEQ/128, BATCH), 256, 0, stream>>>(
        Qb, SD, EMBED, Kb, SD, EMBED,
        scores, (unsigned short*)0, SS, SEQ, (const float*)0, inv_sqrt_d, EMBED, 0);

    // 6) softmax rows -> bf16
    softmax_kernel<<<dim3(BATCH * SEQ), 256, 0, stream>>>(scores, attnb, SEQ);

    // 7) attn_out = P @ V, batched (Bt = Vt [D][S])
    gemm_bf16_wmma<<<dim3(EMBED/128, SEQ/128, BATCH), 256, 0, stream>>>(
        attnb, SS, SEQ, Vt, SD, SEQ,
        attno, (unsigned short*)0, SD, EMBED, (const float*)0, 1.0f, SEQ, 0);

    // 8) x1 = x + LN(attn_out)*g1 + beta1  (f32 + bf16)
    ln_residual_kernel<<<dim3(TOKENS), 256, 0, stream>>>(
        x, attno, g1, bt1, x1, x1b, EMBED);

    // 9) h = gelu(x1 @ W1 + b1) (bf16)
    gemm_bf16_wmma<<<dim3(FFDIM/128, TOKENS/128, 1), 256, 0, stream>>>(
        x1b, 0, EMBED, W1t, 0, EMBED,
        (float*)0, hb, 0, FFDIM, b1, 1.0f, EMBED, 1);

    // 10) ffn = h @ W2 + b2 (f32)
    gemm_bf16_wmma<<<dim3(EMBED/128, TOKENS/128, 1), 256, 0, stream>>>(
        hb, 0, FFDIM, W2t, 0, FFDIM,
        ffn, (unsigned short*)0, 0, EMBED, b2, 1.0f, FFDIM, 0);

    // 11) out = x1 + LN(ffn)*g2 + beta2
    ln_residual_kernel<<<dim3(TOKENS), 256, 0, stream>>>(
        x1, ffn, g2, bt2, out, (unsigned short*)0, EMBED);
}
